// GQAttention_18098992185843
// MI455X (gfx1250) — compile-verified, matrix operand order checked by reading, not run on hardware
//
#include <hip/hip_runtime.h>
#include <hip/hip_bf16.h>
#include <math.h>

// ---------------------------------------------------------------------------
// Problem constants (match reference)
// ---------------------------------------------------------------------------
#define BB        2
#define TT        2048
#define DD        2048
#define NHEADS    16
#define NKVHEADS  4
#define HEADDIM   128
#define DKV       (NKVHEADS * HEADDIM)   // 512
#define MROWS     (BB * TT)              // 4096

typedef __attribute__((ext_vector_type(16))) __bf16 bf16x16;
typedef __attribute__((ext_vector_type(8)))  float  f32x8;
typedef __attribute__((ext_vector_type(4)))  int    v4i_t;

// addrspace-qualified vector types for the async-to-LDS builtin
typedef __attribute__((address_space(1))) v4i_t glob_v4i;
typedef __attribute__((address_space(3))) v4i_t lds_v4i;

union FragU { bf16x16 v; uint4 q[2]; };

// --- CDNA5 async global->LDS path (ASYNCcnt), with sync fallback -----------
#ifndef HAS_ASYNC_LDS
#define HAS_ASYNC_LDS 0
#endif
#ifndef HAS_WAIT_ASYNC
#define HAS_WAIT_ASYNC 0
#endif
#if defined(__has_builtin)
#  if __has_builtin(__builtin_amdgcn_global_load_async_to_lds_b128)
#    undef HAS_ASYNC_LDS
#    define HAS_ASYNC_LDS 1
#  endif
#  if __has_builtin(__builtin_amdgcn_s_wait_asynccnt)
#    undef HAS_WAIT_ASYNC
#    define HAS_WAIT_ASYNC 1
#  endif
#endif

// copy 16 bytes global -> LDS without VGPR transit (GLOBAL_LOAD_ASYNC_TO_LDS_B128)
__device__ __forceinline__ void async_copy_b128(void* lds, const void* g) {
#if HAS_ASYNC_LDS
  __builtin_amdgcn_global_load_async_to_lds_b128(
      (glob_v4i*)g, (lds_v4i*)lds, 0, 0);
#else
  *(uint4*)lds = *(const uint4*)g;
#endif
}

__device__ __forceinline__ void wait_async_all() {
#if HAS_ASYNC_LDS
#  if HAS_WAIT_ASYNC
  __builtin_amdgcn_s_wait_asynccnt(0);
#  else
  asm volatile("s_wait_asynccnt 0x0" ::: "memory");
#  endif
#endif
}

// fp32 -> bf16, round to nearest even
__device__ __forceinline__ __bf16 f2bf(float f) {
  union { float f; unsigned u; } v; v.f = f;
  unsigned r = v.u + 0x7FFFu + ((v.u >> 16) & 1u);
  unsigned short h = (unsigned short)(r >> 16);
  __bf16 b; __builtin_memcpy(&b, &h, 2); return b;
}

__device__ __forceinline__ f32x8 wmma_bf16(bf16x16 a, bf16x16 b, f32x8 c) {
  return __builtin_amdgcn_wmma_f32_16x16x32_bf16(false, a, false, b,
                                                 (short)0, c, false, false);
}

// Load one 16x32 bf16 operand fragment (A-layout) from a row-major tile.
// lane 0-15: rows 0-15, k-halves h*8+0..7 and 16+h*8+0..7 (h = lane>>4).
// base must be 16B aligned, ld (elements)*2 must be a multiple of 16.
__device__ __forceinline__ bf16x16 lds_frag(const __bf16* base, int ld) {
  const int lane = threadIdx.x & 31;
  const int m = lane & 15, h = lane >> 4;
  const __bf16* row = base + m * ld + h * 8;
  FragU f;
  f.q[0] = *(const uint4*)(row);
  f.q[1] = *(const uint4*)(row + 16);
  return f.v;
}

// ---------------------------------------------------------------------------
// fp32 -> bf16 bulk convert (n multiple of 4)
// ---------------------------------------------------------------------------
__global__ void convert4_f32_bf16(const float* __restrict__ in,
                                  __bf16* __restrict__ out, int n4) {
  int i = blockIdx.x * blockDim.x + threadIdx.x;
  if (i >= n4) return;
  float4 v = ((const float4*)in)[i];
  union { __bf16 e[4]; uint2 u; } r;
  r.e[0] = f2bf(v.x); r.e[1] = f2bf(v.y);
  r.e[2] = f2bf(v.z); r.e[3] = f2bf(v.w);
  ((uint2*)out)[i] = r.u;
}

// ---------------------------------------------------------------------------
// C[M,N] (f32) = A[M,K] (bf16, row major) * W[N,K]^T (bf16, row major)
// Block tile 128x128, 8 waves (4x2), each wave 32x64 -> 2x4 WMMA tiles.
// Double-buffered LDS; K-slice copies are async (overlap WMMA compute).
// ---------------------------------------------------------------------------
#define GBM 128
#define GBN 128
#define GBK 32
#define GLD 40   // padded LDS row stride (elements); 80B rows -> 16B aligned

__global__ __launch_bounds__(256) void gemm_bf16_nt(
    const __bf16* __restrict__ A, const __bf16* __restrict__ W,
    float* __restrict__ C, int M, int N, int K) {
  __shared__ __bf16 As[2][GBM * GLD];
  __shared__ __bf16 Ws[2][GBN * GLD];
  const int tid  = threadIdx.x;
  const int lane = tid & 31;
  const int wave = tid >> 5;
  const int wm = wave >> 1, wn = wave & 1;            // 4x2 wave grid
  const int bm = blockIdx.y * GBM, bn = blockIdx.x * GBN;
  const int cn = lane & 15, hh = lane >> 4;

  // each thread owns two 16B chunks of each 128x32 tile (512 uint4 / tile)
  const int rowA0 = tid >> 2,          c80 = (tid & 3) * 8;
  const int rowA1 = (tid + 256) >> 2,  c81 = c80;

  f32x8 acc[2][4] = {};

  auto issue = [&](int buf, int k0) {
    async_copy_b128(&As[buf][rowA0 * GLD + c80],
                    &A[(size_t)(bm + rowA0) * K + k0 + c80]);
    async_copy_b128(&As[buf][rowA1 * GLD + c81],
                    &A[(size_t)(bm + rowA1) * K + k0 + c81]);
    async_copy_b128(&Ws[buf][rowA0 * GLD + c80],
                    &W[(size_t)(bn + rowA0) * K + k0 + c80]);
    async_copy_b128(&Ws[buf][rowA1 * GLD + c81],
                    &W[(size_t)(bn + rowA1) * K + k0 + c81]);
  };

  issue(0, 0);
  const int nk = K / GBK;
  for (int i = 0; i < nk; ++i) {
    const int buf = i & 1;
    wait_async_all();        // this wave's copies for tile i are done
    __syncthreads();         // all waves done copying i and done reading buf^1
    if (i + 1 < nk) issue(buf ^ 1, (i + 1) * GBK);   // overlaps compute below

    bf16x16 af[2], wf[4];
#pragma unroll
    for (int ii = 0; ii < 2; ++ii)
      af[ii] = lds_frag(&As[buf][(wm * 32 + ii * 16) * GLD], GLD);
#pragma unroll
    for (int j = 0; j < 4; ++j)
      wf[j] = lds_frag(&Ws[buf][(wn * 64 + j * 16) * GLD], GLD);
#pragma unroll
    for (int ii = 0; ii < 2; ++ii)
#pragma unroll
      for (int j = 0; j < 4; ++j)
        acc[ii][j] = wmma_bf16(af[ii], wf[j], acc[ii][j]);
  }

  const int m0 = bm + wm * 32;
  const int n0 = bn + wn * 64;
#pragma unroll
  for (int i = 0; i < 2; ++i)
#pragma unroll
    for (int j = 0; j < 4; ++j)
#pragma unroll
      for (int r = 0; r < 8; ++r)
        C[(size_t)(m0 + i * 16 + r + 8 * hh) * N + n0 + j * 16 + cn] =
            acc[i][j][r];
}

// ---------------------------------------------------------------------------
// RoPE on fp32 projection output, packed to bf16.
// X: (MROWS, nheads*128) f32 ; R: (T, 64, 2) f32 {cos,sin} ; Y: bf16 same shape
// ---------------------------------------------------------------------------
__global__ void rope_pack(const float* __restrict__ X, const float* __restrict__ R,
                          __bf16* __restrict__ Y, int nheads) {
  int idx   = blockIdx.x * blockDim.x + threadIdx.x;
  int total = MROWS * nheads * 64;
  if (idx >= total) return;
  int i   = idx & 63;
  int h   = (idx >> 6) % nheads;
  int row = idx / (64 * nheads);
  int t   = row & (TT - 1);
  float c = R[(t * 64 + i) * 2 + 0];
  float s = R[(t * 64 + i) * 2 + 1];
  int base = row * (nheads * HEADDIM) + h * HEADDIM;
  float x1 = X[base + i], x2 = X[base + 64 + i];
  Y[base + i]      = f2bf(x1 * c - x2 * s);
  Y[base + 64 + i] = f2bf(x1 * s + x2 * c);
}

// ---------------------------------------------------------------------------
// Flash attention (causal, GQA). 128-query tile per block, 8 waves x 16 rows.
// KV tiles of 32 keys staged in LDS (K via async DMA); all matmuls on
// v_wmma_f32_16x16x32_bf16.
// Q: (MROWS, 2048) bf16 ; K,V: (MROWS, 512) bf16 ; O: (MROWS, 2048) bf16
// ---------------------------------------------------------------------------
#define FBQ 128
#define FBK 32
#define KLD 136   // 272B rows
#define VLD 40    // 80B rows

__global__ __launch_bounds__(256) void flash_fwd(
    const __bf16* __restrict__ Q, const __bf16* __restrict__ K,
    const __bf16* __restrict__ V, __bf16* __restrict__ O) {
  __shared__ __bf16 Ks[FBK * KLD];        // [key][dim]
  __shared__ __bf16 Vt[HEADDIM * VLD];    // [dim][key] (transposed)
  __shared__ __bf16 Ps[8 * 16 * VLD];     // per-wave P staging [row][key]

  const int tid  = threadIdx.x;
  const int lane = tid & 31;
  const int wave = tid >> 5;
  const int cn = lane & 15, hh = lane >> 4;
  const int b = blockIdx.z, h = blockIdx.y;
  const int kvh = h >> 2;                 // N_GROUPS = 4
  const int q0 = blockIdx.x * FBQ;
  const int qrowg = q0 + wave * 16;

  // Q fragments for this wave's 16 rows: 4 k-chunks of 32
  bf16x16 qf[4];
  {
    const __bf16* qrow =
        Q + (size_t)(b * TT + qrowg + cn) * DD + h * HEADDIM + hh * 8;
    FragU f;
#pragma unroll
    for (int c = 0; c < 4; ++c) {
      f.q[0] = *(const uint4*)(qrow + c * 32);
      f.q[1] = *(const uint4*)(qrow + c * 32 + 16);
      qf[c] = f.v;
    }
  }

  f32x8 oacc[8] = {};
  float m_i[8], l_i[8];
#pragma unroll
  for (int j = 0; j < 8; ++j) { m_i[j] = -__builtin_inff(); l_i[j] = 0.f; }
  const float scale = 0.08838834764831845f;   // 1/sqrt(128)

  const int kv_end = q0 + FBQ;
  for (int kv0 = 0; kv0 < kv_end; kv0 += FBK) {
    // --- K tile: 32 keys x 128 dims via async global->LDS DMA
#pragma unroll
    for (int r = 0; r < 2; ++r) {
      int u   = tid + r * 256;
      int key = u >> 4;
      int c8  = (u & 15) * 8;
      async_copy_b128(&Ks[key * KLD + c8],
                      &K[(size_t)(b * TT + kv0 + key) * DKV +
                         kvh * HEADDIM + c8]);
    }
    // --- V tile load + transpose: Vt[dim][key] (2-byte scatter, via VGPRs)
    {
      int key = tid >> 3;
      int d0  = (tid & 7) * 16;
      union { uint4 q[2]; __bf16 e[16]; } tv;
      const __bf16* src =
          &V[(size_t)(b * TT + kv0 + key) * DKV + kvh * HEADDIM + d0];
      tv.q[0] = *(const uint4*)src;
      tv.q[1] = *(const uint4*)(src + 8);
#pragma unroll
      for (int jj = 0; jj < 16; ++jj)
        Vt[(d0 + jj) * VLD + key] = tv.e[jj];
    }
    wait_async_all();
    __syncthreads();

    // --- S = Q K^T : two 16x16 score tiles, 4 k-chunks each
    f32x8 s[2] = {};
#pragma unroll
    for (int c = 0; c < 4; ++c) {
#pragma unroll
      for (int nt = 0; nt < 2; ++nt) {
        bf16x16 kf = lds_frag(&Ks[nt * 16 * KLD + c * 32], KLD);
        s[nt] = wmma_bf16(qf[c], kf, s[nt]);
      }
    }

    // --- causal mask + online softmax (rows span lanes 0-15 / 16-31)
    float p0a[8], p1a[8], corr[8];
#pragma unroll
    for (int j = 0; j < 8; ++j) {
      int qr = qrowg + j + 8 * hh;
      float v0 = s[0][j] * scale;
      float v1 = s[1][j] * scale;
      if (kv0 + cn > qr)      v0 = -__builtin_inff();
      if (kv0 + 16 + cn > qr) v1 = -__builtin_inff();
      float rm = fmaxf(v0, v1);
#pragma unroll
      for (int off = 8; off >= 1; off >>= 1)
        rm = fmaxf(rm, __shfl_xor(rm, off, 32));
      float mo = m_i[j];
      float mn = fmaxf(mo, rm);
      float c0 = __expf(mo - mn);
      float p0 = (v0 == -__builtin_inff()) ? 0.f : __expf(v0 - mn);
      float p1 = (v1 == -__builtin_inff()) ? 0.f : __expf(v1 - mn);
      float rs = p0 + p1;
#pragma unroll
      for (int off = 8; off >= 1; off >>= 1)
        rs += __shfl_xor(rs, off, 32);
      l_i[j] = l_i[j] * c0 + rs;
      m_i[j] = mn;
      corr[j] = c0;
      p0a[j] = p0; p1a[j] = p1;
    }

    // --- rescale O accumulators
#pragma unroll
    for (int t = 0; t < 8; ++t)
#pragma unroll
      for (int j = 0; j < 8; ++j)
        oacc[t][j] *= corr[j];

    // --- stage P (C-layout) -> LDS -> reload as A fragment (per-wave region)
    __bf16* pst = &Ps[wave * 16 * VLD];
#pragma unroll
    for (int j = 0; j < 8; ++j) {
      pst[(j + 8 * hh) * VLD + cn]      = f2bf(p0a[j]);
      pst[(j + 8 * hh) * VLD + 16 + cn] = f2bf(p1a[j]);
    }
    asm volatile("s_wait_dscnt 0" ::: "memory");
    bf16x16 pf = lds_frag(pst, VLD);

    // --- O += P V  (8 dim tiles of 16)
#pragma unroll
    for (int t = 0; t < 8; ++t) {
      bf16x16 vf = lds_frag(&Vt[t * 16 * VLD], VLD);
      oacc[t] = wmma_bf16(pf, vf, oacc[t]);
    }
    __syncthreads();
  }

  // --- epilogue: normalize, pack bf16
  float inv[8];
#pragma unroll
  for (int j = 0; j < 8; ++j) inv[j] = 1.0f / l_i[j];
#pragma unroll
  for (int t = 0; t < 8; ++t)
#pragma unroll
    for (int j = 0; j < 8; ++j) {
      int row = b * TT + qrowg + j + 8 * hh;
      int col = h * HEADDIM + t * 16 + cn;
      O[(size_t)row * DD + col] = f2bf(oacc[t][j] * inv[j]);
    }
}

// ---------------------------------------------------------------------------
// Host orchestration
// ---------------------------------------------------------------------------
extern "C" void kernel_launch(void* const* d_in, const int* in_sizes, int n_in,
                              void* d_out, int out_size, void* d_ws, size_t ws_size,
                              hipStream_t stream) {
  (void)in_sizes; (void)n_in; (void)out_size; (void)ws_size;
  const float* x    = (const float*)d_in[0];
  const float* rope = (const float*)d_in[1];
  const float* wq   = (const float*)d_in[2];
  const float* wk   = (const float*)d_in[3];
  const float* wv   = (const float*)d_in[4];
  const float* wo   = (const float*)d_in[5];
  float* out = (float*)d_out;

  char* ws = (char*)d_ws;
  size_t off = 0;
  auto alloc = [&](size_t bytes) -> void* {
    void* p = ws + off;
    off = (off + bytes + 255) & ~(size_t)255;
    return p;
  };

  __bf16* xb  = (__bf16*)alloc((size_t)MROWS * DD * 2);
  __bf16* wqb = (__bf16*)alloc((size_t)DD * DD * 2);
  __bf16* wkb = (__bf16*)alloc((size_t)DKV * DD * 2);
  __bf16* wvb = (__bf16*)alloc((size_t)DKV * DD * 2);
  __bf16* wob = (__bf16*)alloc((size_t)DD * DD * 2);
  float*  Qf  = (float*)alloc((size_t)MROWS * DD * 4);
  float*  Kf  = (float*)alloc((size_t)MROWS * DKV * 4);
  float*  Vf  = (float*)alloc((size_t)MROWS * DKV * 4);
  __bf16* Qb  = (__bf16*)alloc((size_t)MROWS * DD * 2);
  __bf16* Kb  = (__bf16*)alloc((size_t)MROWS * DKV * 2);
  __bf16* Vb  = (__bf16*)alloc((size_t)MROWS * DKV * 2);
  __bf16* Ob  = (__bf16*)alloc((size_t)MROWS * DD * 2);

  auto cvt = [&](const float* src, __bf16* dst, size_t n) {
    int n4 = (int)(n / 4);
    convert4_f32_bf16<<<(n4 + 255) / 256, 256, 0, stream>>>(src, dst, n4);
  };

  // 1) bf16 conversions
  cvt(x,  xb,  (size_t)MROWS * DD);
  cvt(wq, wqb, (size_t)DD * DD);
  cvt(wk, wkb, (size_t)DKV * DD);
  cvt(wv, wvb, (size_t)DKV * DD);
  cvt(wo, wob, (size_t)DD * DD);

  // 2) QKV projections (WMMA GEMMs)
  gemm_bf16_nt<<<dim3(DD / GBN,  MROWS / GBM), 256, 0, stream>>>(
      xb, wqb, Qf, MROWS, DD, DD);
  gemm_bf16_nt<<<dim3(DKV / GBN, MROWS / GBM), 256, 0, stream>>>(
      xb, wkb, Kf, MROWS, DKV, DD);
  gemm_bf16_nt<<<dim3(DKV / GBN, MROWS / GBM), 256, 0, stream>>>(
      xb, wvb, Vf, MROWS, DKV, DD);

  // 3) RoPE (Q, K) + pack; V pack
  {
    int totq = MROWS * NHEADS * 64;
    rope_pack<<<(totq + 255) / 256, 256, 0, stream>>>(Qf, rope, Qb, NHEADS);
    int totk = MROWS * NKVHEADS * 64;
    rope_pack<<<(totk + 255) / 256, 256, 0, stream>>>(Kf, rope, Kb, NKVHEADS);
    cvt(Vf, Vb, (size_t)MROWS * DKV);
  }

  // 4) flash attention (WMMA)
  flash_fwd<<<dim3(TT / FBQ, NHEADS, BB), 256, 0, stream>>>(Qb, Kb, Vb, Ob);

  // 5) output projection (WMMA GEMM) -> fp32 out
  gemm_bf16_nt<<<dim3(DD / GBN, MROWS / GBM), 256, 0, stream>>>(
      Ob, wob, out, MROWS, DD, DD);
}
